// HandDecoder_nerf2d_7189775253537
// MI455X (gfx1250) — matrix operand order repaired and checked by reading, not compiled, not run on hardware
//
#include <hip/hip_runtime.h>
#include <hip/hip_bf16.h>

// Problem constants (match reference).
#define NB 8
#define NP 196
#define C0 64
#define PEW 60
#define XDIM (C0 + PEW)   // 124
#define C1 32
#define C2 16
#define NTILE 13          // ceil(196/16)
#define NJP   98          // 196/2 j-pairs (K=32 per WMMA)

typedef __attribute__((ext_vector_type(16))) _Float16 v16h;
typedef __attribute__((ext_vector_type(8)))  float    v8f;
typedef __attribute__((ext_vector_type(2)))  _Float16 h2v;   // packed f16 pair

__device__ __forceinline__ h2v h2splat(float v) {
  _Float16 e = (_Float16)v;
  h2v r = {e, e};
  return r;
}
__device__ __forceinline__ h2v h2relu(h2v v) {
  h2v z = {(_Float16)0.0f, (_Float16)0.0f};
  return __builtin_elementwise_max(v, z);   // v_pk_max_num_f16
}

// ---------------------------------------------------------------------------
// Kernel 1: per-node decode MLP -> feat (packed f16 VOP3P math), then
// G[b,j,t,d] = feat @ kW3, stored transposed as Gt[b][j][d][t] in f16 so WMMA
// B-fragments are two contiguous 16B loads per lane. kW3 is staged in LDS
// pre-transposed+packed as [c][d][t-pair] half2, so G accumulators are born in
// the packed [d][t] store layout. kb3 bias term accumulated in f32 atomics.
// ---------------------------------------------------------------------------
__global__ __launch_bounds__(256)
void nerf_node_stage(const float* __restrict__ feature,
                     const float* __restrict__ pe,
                     const float* __restrict__ dW1, const float* __restrict__ db1,
                     const float* __restrict__ dW2, const float* __restrict__ db2,
                     const float* __restrict__ dW3, const float* __restrict__ db3,
                     const float* __restrict__ kW3, const float* __restrict__ kb3,
                     _Float16* __restrict__ Gt, float* __restrict__ biasbuf) {
  __shared__ h2v   s_dW1h[XDIM * 16];      // [i][op]  op = output pair
  __shared__ h2v   s_dW2h[32 * 8];         // [i][op]
  __shared__ h2v   s_dW3h[16 * 16];        // [i][op]
  __shared__ h2v   s_kW3h[C1 * C2 * 8];    // [c][d][tp]  (transposed, packed t)
  __shared__ h2v   s_db1p[16];
  __shared__ h2v   s_db2p[8];
  __shared__ h2v   s_db3p[16];
  __shared__ float s_kb3[C1 * C2];

  const int t = threadIdx.x;
  for (int k = t; k < XDIM * 16; k += 256) {
    const int i = k >> 4, op = k & 15;
    s_dW1h[k] = (h2v){(_Float16)dW1[i * 32 + 2 * op], (_Float16)dW1[i * 32 + 2 * op + 1]};
  }
  for (int k = t; k < 32 * 8; k += 256) {
    const int i = k >> 3, op = k & 7;
    s_dW2h[k] = (h2v){(_Float16)dW2[i * 16 + 2 * op], (_Float16)dW2[i * 16 + 2 * op + 1]};
  }
  for (int k = t; k < 16 * 16; k += 256) {
    const int i = k >> 4, op = k & 15;
    s_dW3h[k] = (h2v){(_Float16)dW3[i * 32 + 2 * op], (_Float16)dW3[i * 32 + 2 * op + 1]};
  }
  for (int k = t; k < C1 * C2 * 8; k += 256) {   // k = (c*16+d)*8+tp
    const int tp = k & 7, d = (k >> 3) & 15, c = k >> 7;
    s_kW3h[k] = (h2v){(_Float16)kW3[(2 * tp)     * (C1 * C2) + c * C2 + d],
                      (_Float16)kW3[(2 * tp + 1) * (C1 * C2) + c * C2 + d]};
  }
  for (int k = t; k < C1 * C2; k += 256) s_kb3[k] = kb3[k];
  if (t < 16) s_db1p[t] = (h2v){(_Float16)db1[2 * t], (_Float16)db1[2 * t + 1]};
  if (t < 8)  s_db2p[t] = (h2v){(_Float16)db2[2 * t], (_Float16)db2[2 * t + 1]};
  if (t < 16) s_db3p[t] = (h2v){(_Float16)db3[2 * t], (_Float16)db3[2 * t + 1]};
  __syncthreads();

  const int idx = blockIdx.x * 256 + t;
  if (idx >= NB * NP) return;
  const int b = idx / NP;
  const int p = idx - b * NP;

  // x = concat(feature[b], pe[b,p]) as broadcast f16 pairs.
  h2v xh[XDIM];
  #pragma unroll 4
  for (int c = 0; c < C0; ++c)  xh[c]      = h2splat(feature[b * C0 + c]);
  #pragma unroll 4
  for (int q = 0; q < PEW; ++q) xh[C0 + q] = h2splat(pe[(size_t)(b * NP + p) * PEW + q]);

  // Layer 1: 124 -> 32 (16 packed accumulators).
  h2v h1p[16];
  #pragma unroll
  for (int op = 0; op < 16; ++op) h1p[op] = s_db1p[op];
  #pragma unroll 4
  for (int i = 0; i < XDIM; ++i) {
    const h2v xv = xh[i];
    #pragma unroll
    for (int op = 0; op < 16; ++op) h1p[op] += xv * s_dW1h[i * 16 + op];
  }
  h2v h1s[32];
  #pragma unroll
  for (int i = 0; i < 32; ++i) {
    h2v r = h2relu(h1p[i >> 1]);
    _Float16 e = r[i & 1];
    h1s[i] = (h2v){e, e};
  }

  // Layer 2: 32 -> 16.
  h2v h2p[8];
  #pragma unroll
  for (int op = 0; op < 8; ++op) h2p[op] = s_db2p[op];
  #pragma unroll
  for (int i = 0; i < 32; ++i) {
    #pragma unroll
    for (int op = 0; op < 8; ++op) h2p[op] += h1s[i] * s_dW2h[i * 8 + op];
  }
  h2v h2s[16];
  #pragma unroll
  for (int i = 0; i < 16; ++i) {
    h2v r = h2relu(h2p[i >> 1]);
    _Float16 e = r[i & 1];
    h2s[i] = (h2v){e, e};
  }

  // Layer 3: 16 -> 32 (feat).
  h2v fp[16];
  #pragma unroll
  for (int op = 0; op < 16; ++op) fp[op] = s_db3p[op];
  #pragma unroll
  for (int i = 0; i < 16; ++i) {
    #pragma unroll
    for (int op = 0; op < 16; ++op) fp[op] += h2s[i] * s_dW3h[i * 16 + op];
  }
  h2v fs[C1];
  #pragma unroll
  for (int c = 0; c < C1; ++c) {
    h2v r = h2relu(fp[c >> 1]);
    _Float16 e = r[c & 1];
    fs[c] = (h2v){e, e};
  }

  // G expansion: Gt[b][p][d][t] = sum_c f[c]*kW3[t][c*16+d]; packed over t.
  uint4* gout = (uint4*)(Gt + ((size_t)(b * NP + p) * C2) * 16);
  for (int d = 0; d < C2; ++d) {
    union { h2v a[8]; uint4 q[2]; } u;
    #pragma unroll
    for (int tp = 0; tp < 8; ++tp) u.a[tp] = (h2v){(_Float16)0.0f, (_Float16)0.0f};
    #pragma unroll 4
    for (int c = 0; c < C1; ++c) {
      #pragma unroll
      for (int tp = 0; tp < 8; ++tp) u.a[tp] += fs[c] * s_kW3h[(c * C2 + d) * 8 + tp];
    }
    gout[d * 2 + 0] = u.q[0];
    gout[d * 2 + 1] = u.q[1];
  }

  // bias[b][d] += sum_c f[c]*kb3[c*16+d]  (i-independent term, f32 atomics).
  for (int d = 0; d < C2; ++d) {
    float s = 0.f;
    #pragma unroll
    for (int c = 0; c < C1; ++c) {
      h2v r = h2relu(fp[c >> 1]);
      s = fmaf((float)r[c & 1], s_kb3[c * C2 + d], s);
    }
    atomicAdd(&biasbuf[b * C2 + d], s);
  }
}

// ---------------------------------------------------------------------------
// Kernel 2 helpers.
// ---------------------------------------------------------------------------
struct Pk2 { uint4 q0, q1; };  // 16 f16 packed

// 3->8->16 kernel MLP for one relative coordinate; layer 2 in packed f16
// (v_pk_fma_f16) with register-resident packed weights. Result is the lane's
// 32B A-tile slice, already f16-packed.
__device__ __forceinline__ Pk2 h2_eval(float rx, float ry, float rz,
                                       const float* __restrict__ w1,
                                       const float* __restrict__ b1,
                                       const h2v* __restrict__ w2p,
                                       const h2v* __restrict__ b2p) {
  float a1[8];
  #pragma unroll
  for (int o = 0; o < 8; ++o)
    a1[o] = fmaxf(fmaf(rz, w1[16 + o], fmaf(ry, w1[8 + o], fmaf(rx, w1[o], b1[o]))), 0.f);
  h2v a1s[8];
  #pragma unroll
  for (int o = 0; o < 8; ++o) a1s[o] = h2splat(a1[o]);

  union { h2v a[8]; Pk2 p; } u;
  #pragma unroll
  for (int tp = 0; tp < 8; ++tp) u.a[tp] = b2p[tp];
  #pragma unroll
  for (int o = 0; o < 8; ++o) {
    #pragma unroll
    for (int tp = 0; tp < 8; ++tp) u.a[tp] += a1s[o] * w2p[o * 8 + tp];
  }
  #pragma unroll
  for (int tp = 0; tp < 8; ++tp) u.a[tp] = h2relu(u.a[tp]);
  return u.p;
}

// ---------------------------------------------------------------------------
// Kernel 2: one wave32 per (batch, 16-row i-tile). 2-way unrolled j-pair loop
// with double-buffered A tile in LDS: per WMMA -> 2 ds_store_b128 +
// 2 ds_load_b128 + 2 global b128 (B frag) + 1 barrier. Weights in VGPRs.
// ---------------------------------------------------------------------------
__global__ __launch_bounds__(32)
void nerf_conv_wmma(const float* __restrict__ coords,
                    const float* __restrict__ kW1, const float* __restrict__ kb1,
                    const float* __restrict__ kW2, const float* __restrict__ kb2,
                    const _Float16* __restrict__ Gt,
                    const float* __restrict__ biasbuf,
                    float* __restrict__ out) {
  __shared__ _Float16 Alds[2][16 * 32];   // double-buffered A tile [M=16][K=32]

  const int lane = threadIdx.x;        // 0..31 (wave32)
  const int blk  = blockIdx.x;         // 0..NB*NTILE-1
  const int b    = blk / NTILE;
  const int i0   = (blk - b * NTILE) * 16;
  const int m    = lane & 15;          // row within tile / N column
  const int h    = lane >> 4;          // half-wave selector

  // Register-resident kernel-MLP weights (layer2 packed f16).
  float w1[3 * 8], b1[8];
  #pragma unroll
  for (int i = 0; i < 24; ++i) w1[i] = kW1[i];
  #pragma unroll
  for (int i = 0; i < 8; ++i)  b1[i] = kb1[i];
  h2v w2p[64], b2p[8];
  #pragma unroll
  for (int k = 0; k < 64; ++k) {
    const int u = k >> 3, tp = k & 7;
    w2p[k] = (h2v){(_Float16)kW2[u * 16 + 2 * tp], (_Float16)kW2[u * 16 + 2 * tp + 1]};
  }
  #pragma unroll
  for (int tp = 0; tp < 8; ++tp)
    b2p[tp] = (h2v){(_Float16)kb2[2 * tp], (_Float16)kb2[2 * tp + 1]};

  const int irow = min(i0 + m, NP - 1);           // clamp padded rows
  const float cix = coords[(size_t)(b * NP + irow) * 3 + 0];
  const float ciy = coords[(size_t)(b * NP + irow) * 3 + 1];
  const float ciz = coords[(size_t)(b * NP + irow) * 3 + 2];

  // Lane's 32B h2 slice lives at halves [m*32 + h*16, +16) -> one uint4 pair.
  uint4* a_st0 = (uint4*)&Alds[0][m * 32 + h * 16];
  uint4* a_st1 = (uint4*)&Alds[1][m * 32 + h * 16];
  // A-fragment = two contiguous 4-dword runs: u32 idx m*16+8h and m*16+8h+4.
  const uint4* a_ld0 = (const uint4*)&Alds[0][0];
  const uint4* a_ld1 = (const uint4*)&Alds[1][0];
  const int fi0 = m * 4 + 2 * h;
  const int fi1 = m * 4 + 2 * h + 1;

  const float* cb = coords + (size_t)b * NP * 3;
  const _Float16* gb = Gt + (size_t)b * NP * C2 * 16;

  v8f acc = {};
  // Prologue: stage tile for jp=0 into buffer 0.
  {
    const int j = h;                                  // 2*0 + h
    Pk2 p = h2_eval(cb[j * 3 + 0] - cix, cb[j * 3 + 1] - ciy, cb[j * 3 + 2] - ciz,
                    w1, b1, w2p, b2p);
    a_st0[0] = p.q0;
    a_st0[1] = p.q1;
  }

  for (int jp = 0; jp < NJP; jp += 2) {
    __syncthreads();                                  // buffer 0 tile visible

    // ---- sub-iteration A: consume buffer 0 (jp), produce buffer 1 (jp+1) ---
    union { uint4 q[2]; v16h v; } af0;
    af0.q[0] = a_ld0[fi0];
    af0.q[1] = a_ld0[fi1];
    union { uint4 q[2]; v16h v; } bf0;
    {
      const uint4* gq = (const uint4*)(gb + ((size_t)(2 * jp + h) * C2 + m) * 16);
      bf0.q[0] = gq[0];
      bf0.q[1] = gq[1];
    }
    __builtin_prefetch(gb + ((size_t)(2 * jp + 2 + h) * C2 + m) * 16, 0, 0);
    {
      const int j = 2 * jp + 2 + h;                   // jp+1's column
      Pk2 p = h2_eval(cb[j * 3 + 0] - cix, cb[j * 3 + 1] - ciy, cb[j * 3 + 2] - ciz,
                      w1, b1, w2p, b2p);
      a_st1[0] = p.q0;
      a_st1[1] = p.q1;
    }
    acc = __builtin_amdgcn_wmma_f32_16x16x32_f16(
        false, af0.v, false, bf0.v, (short)0, acc, false, false);

    __syncthreads();                                  // buffer 1 tile visible

    // ---- sub-iteration B: consume buffer 1 (jp+1), produce buffer 0 (jp+2) -
    union { uint4 q[2]; v16h v; } af1;
    af1.q[0] = a_ld1[fi0];
    af1.q[1] = a_ld1[fi1];
    union { uint4 q[2]; v16h v; } bf1;
    {
      const uint4* gq = (const uint4*)(gb + ((size_t)(2 * jp + 2 + h) * C2 + m) * 16);
      bf1.q[0] = gq[0];
      bf1.q[1] = gq[1];
    }
    if (jp + 2 < NJP) {
      __builtin_prefetch(gb + ((size_t)(2 * jp + 4 + h) * C2 + m) * 16, 0, 0);
      const int j = 2 * jp + 4 + h;                   // jp+2's column
      Pk2 p = h2_eval(cb[j * 3 + 0] - cix, cb[j * 3 + 1] - ciy, cb[j * 3 + 2] - ciz,
                      w1, b1, w2p, b2p);
      a_st0[0] = p.q0;
      a_st0[1] = p.q1;
    }
    acc = __builtin_amdgcn_wmma_f32_16x16x32_f16(
        false, af1.v, false, bf1.v, (short)0, acc, false, false);
  }

  // Epilogue: C layout -> row M = r + 8h, col N = m. Add bias, ReLU, store.
  const float bb = biasbuf[b * C2 + m];
  #pragma unroll
  for (int r = 0; r < 8; ++r) {
    const int i = i0 + r + 8 * h;
    if (i < NP) out[((size_t)(b * NP + i)) * C2 + m] = fmaxf(acc[r] + bb, 0.f);
  }
}

// ---------------------------------------------------------------------------
extern "C" void kernel_launch(void* const* d_in, const int* in_sizes, int n_in,
                              void* d_out, int out_size, void* d_ws, size_t ws_size,
                              hipStream_t stream) {
  const float* feature = (const float*)d_in[0];
  const float* pe      = (const float*)d_in[1];
  const float* coords  = (const float*)d_in[2];
  const float* dW1 = (const float*)d_in[3];
  const float* db1 = (const float*)d_in[4];
  const float* dW2 = (const float*)d_in[5];
  const float* db2 = (const float*)d_in[6];
  const float* dW3 = (const float*)d_in[7];
  const float* db3 = (const float*)d_in[8];
  const float* kW1 = (const float*)d_in[9];
  const float* kb1 = (const float*)d_in[10];
  const float* kW2 = (const float*)d_in[11];
  const float* kb2 = (const float*)d_in[12];
  const float* kW3 = (const float*)d_in[13];
  const float* kb3 = (const float*)d_in[14];

  // Workspace: Gt (f16, [B][P][16][16]) then bias (f32, [B][16]).
  const size_t gt_bytes = (size_t)NB * NP * C2 * 16 * sizeof(_Float16); // 802816
  _Float16* Gt      = (_Float16*)d_ws;
  float*    biasbuf = (float*)((char*)d_ws + gt_bytes);

  hipMemsetAsync(biasbuf, 0, NB * C2 * sizeof(float), stream);

  const int nodes = NB * NP;
  nerf_node_stage<<<(nodes + 255) / 256, 256, 0, stream>>>(
      feature, pe, dW1, db1, dW2, db2, dW3, db3, kW3, kb3, Gt, biasbuf);

  nerf_conv_wmma<<<NB * NTILE, 32, 0, stream>>>(
      coords, kW1, kb1, kW2, kb2, Gt, biasbuf, (float*)d_out);
}
